// RecurrentEncoder_19516331393697
// MI455X (gfx1250) — matrix operand order repaired and checked by reading, not compiled, not run on hardware
//
#include <hip/hip_runtime.h>
#include <hip/hip_bf16.h>
#include <math.h>

typedef __bf16 bf16;
typedef __attribute__((ext_vector_type(16))) __bf16 v16bf;
typedef __attribute__((ext_vector_type(8)))  __bf16 v8bf;
typedef __attribute__((ext_vector_type(8)))  float  v8f;

#define Bb 4
#define Nn 1024
#define Dd 1024
#define Hh 16
#define DKk 64
#define DVv 64
#define DIi 4096
#define Tt 4

// ---------------------------------------------------------------- WMMA core
__device__ __forceinline__ v8f wmma_bf16(v16bf a, v16bf b, v8f c) {
  // D = A(16x32) * B(32x16) + C, f32 accumulate
  return __builtin_amdgcn_wmma_f32_16x16x32_bf16(false, a, false, b, (short)0, c, false, false);
}

// A fragment: row-major A[M][K], lane L: row M=L%16,
// elems j = A[M][k0 + j], k0 = kb + (L>=16 ? 16 : 0)
__device__ __forceinline__ v16bf load_frag_a(const bf16* A, int lda, int mb, int kb) {
  const int lane = threadIdx.x & 31;
  const int row  = mb + (lane & 15);
  const int k0   = kb + ((lane & 16) ? 16 : 0);
  const bf16* p = A + (size_t)row * lda + k0;
  return *(const v16bf*)(p);
}

// B fragment from K-contiguous [Nout][K] matrix. lane L: col N=L%16,
// elems j = W[N][kb + (L>=16?16:0) + j]  (one contiguous 32B load)
__device__ __forceinline__ v16bf load_frag_b(const bf16* Wt, int ldb, int nb, int kb) {
  const int lane = threadIdx.x & 31;
  const int col  = nb + (lane & 15);
  const int k0   = kb + ((lane & 16) ? 16 : 0);
  return *(const v16bf*)(Wt + (size_t)col * ldb + k0);
}

struct Acc { v8f c[2][4]; };

// one wave computes a 32x64 f32 tile of A[M][K] x W[Nout][K]^T
__device__ __forceinline__ void wave_gemm(const bf16* __restrict__ A, int lda,
                                          const bf16* __restrict__ Wt, int ldw,
                                          int Kdim, int mb, int nb, Acc& acc) {
  v8f z = {0.f,0.f,0.f,0.f,0.f,0.f,0.f,0.f};
#pragma unroll
  for (int i = 0; i < 2; ++i)
#pragma unroll
    for (int j = 0; j < 4; ++j) acc.c[i][j] = z;
  for (int kb = 0; kb < Kdim; kb += 32) {
    v16bf a0 = load_frag_a(A, lda, mb,      kb);
    v16bf a1 = load_frag_a(A, lda, mb + 16, kb);
#pragma unroll
    for (int j = 0; j < 4; ++j) {
      v16bf b = load_frag_b(Wt, ldw, nb + j*16, kb);
      acc.c[0][j] = wmma_bf16(a0, b, acc.c[0][j]);
      acc.c[1][j] = wmma_bf16(a1, b, acc.c[1][j]);
    }
  }
}

// ------------------------------------------------------------ small kernels
__global__ void cvt_kernel(const float* __restrict__ s, bf16* __restrict__ d,
                           int n, float scale) {
  int i = blockIdx.x * 256 + threadIdx.x;
  if (i < n) d[i] = (bf16)(s[i] * scale);
}

// h = carry + PT(t); a = LN(h) (bf16). One block per row (B*N rows).
__global__ __launch_bounds__(256) void pe_ln_kernel(
    const float* __restrict__ carry, const float* __restrict__ g,
    const float* __restrict__ be, float* __restrict__ hout,
    bf16* __restrict__ aout, int tstep) {
  __shared__ float red1[256], red2[256];
  const int row = blockIdx.x;
  const int n   = row & (Nn - 1);
  const int tid = threadIdx.x;
  float hv[4];
  float lsum = 0.f, lsq = 0.f;
#pragma unroll
  for (int kk = 0; kk < 4; ++kk) {
    int d = tid + kk * 256;
    float f  = __expf(-(float)(d & ~1) * (9.210340371976184f / (float)Dd));
    float pe = (d & 1) ? (cosf((float)n * f) + cosf((float)tstep * f))
                       : (sinf((float)n * f) + sinf((float)tstep * f));
    float v = carry[(size_t)row * Dd + d] + pe;
    hv[kk] = v;
    hout[(size_t)row * Dd + d] = v;
    lsum += v; lsq += v * v;
  }
  red1[tid] = lsum; red2[tid] = lsq;
  __syncthreads();
  for (int s = 128; s > 0; s >>= 1) {
    if (tid < s) { red1[tid] += red1[tid + s]; red2[tid] += red2[tid + s]; }
    __syncthreads();
  }
  float mean = red1[0] * (1.f / Dd);
  float var  = red2[0] * (1.f / Dd) - mean * mean;
  float inv  = rsqrtf(var + 1e-5f);
#pragma unroll
  for (int kk = 0; kk < 4; ++kk) {
    int d = tid + kk * 256;
    aout[(size_t)row * Dd + d] = (bf16)((hv[kk] - mean) * inv * g[d] + be[d]);
  }
}

// r = LN(LN(h2, g1,b1), g2,b2) in bf16
__global__ __launch_bounds__(256) void double_ln_kernel(
    const float* __restrict__ h2,
    const float* __restrict__ g1, const float* __restrict__ b1,
    const float* __restrict__ g2, const float* __restrict__ b2,
    bf16* __restrict__ rout) {
  __shared__ float red1[256], red2[256];
  const int row = blockIdx.x;
  const int tid = threadIdx.x;
  float v[4];
  float lsum = 0.f, lsq = 0.f;
#pragma unroll
  for (int kk = 0; kk < 4; ++kk) {
    int d = tid + kk * 256;
    v[kk] = h2[(size_t)row * Dd + d];
    lsum += v[kk]; lsq += v[kk] * v[kk];
  }
  red1[tid] = lsum; red2[tid] = lsq;
  __syncthreads();
  for (int s = 128; s > 0; s >>= 1) {
    if (tid < s) { red1[tid] += red1[tid + s]; red2[tid] += red2[tid + s]; }
    __syncthreads();
  }
  float mean = red1[0] * (1.f / Dd);
  float var  = red2[0] * (1.f / Dd) - mean * mean;
  float inv  = rsqrtf(var + 1e-5f);
  __syncthreads();
  float y[4];
  lsum = 0.f; lsq = 0.f;
#pragma unroll
  for (int kk = 0; kk < 4; ++kk) {
    int d = tid + kk * 256;
    y[kk] = (v[kk] - mean) * inv * g1[d] + b1[d];
    lsum += y[kk]; lsq += y[kk] * y[kk];
  }
  red1[tid] = lsum; red2[tid] = lsq;
  __syncthreads();
  for (int s = 128; s > 0; s >>= 1) {
    if (tid < s) { red1[tid] += red1[tid + s]; red2[tid] += red2[tid + s]; }
    __syncthreads();
  }
  float mean2 = red1[0] * (1.f / Dd);
  float var2  = red2[0] * (1.f / Dd) - mean2 * mean2;
  float inv2  = rsqrtf(var2 + 1e-5f);
#pragma unroll
  for (int kk = 0; kk < 4; ++kk) {
    int d = tid + kk * 256;
    rout[(size_t)row * Dd + d] = (bf16)((y[kk] - mean2) * inv2 * g2[d] + b2[d]);
  }
}

// ------------------------------------------------------------- GEMM kernels
// QKV: [4096,1024] x Wqkv[3072,1024]^T.  Q scale (1/8) pre-folded into Wq.
// Q,K written [b,h,n,dk]; V written transposed [b,h,dv,n].
__global__ __launch_bounds__(128) void gemm_qkv_kernel(
    const bf16* __restrict__ a, const bf16* __restrict__ wqkv,
    bf16* __restrict__ q, bf16* __restrict__ kk, bf16* __restrict__ vt) {
  const int wid = blockIdx.x * 4 + (threadIdx.x >> 5);
  const int tm = wid & 127, tn = wid >> 7;
  const int mb = tm * 32, nb = tn * 64;
  Acc acc;
  wave_gemm(a, Dd, wqkv, Dd, Dd, mb, nb, acc);
  const int lane = threadIdx.x & 31;
  const int roff = (lane & 16) ? 8 : 0, ncol = lane & 15;
#pragma unroll
  for (int i = 0; i < 2; ++i)
#pragma unroll
    for (int j = 0; j < 4; ++j)
#pragma unroll
      for (int r = 0; r < 8; ++r) {
        int row = mb + i * 16 + r + roff;
        int col = nb + j * 16 + ncol;
        bf16 val = (bf16)acc.c[i][j][r];
        int bidx = row >> 10, ns = row & (Nn - 1);
        if (col < 1024) {
          int hh = col >> 6, kd = col & 63;
          q[((size_t)(bidx * Hh + hh) * Nn + ns) * DKk + kd] = val;
        } else if (col < 2048) {
          int c2 = col - 1024, hh = c2 >> 6, kd = c2 & 63;
          kk[((size_t)(bidx * Hh + hh) * Nn + ns) * DKk + kd] = val;
        } else {
          int c2 = col - 2048, hh = c2 >> 6, kd = c2 & 63;
          vt[((size_t)(bidx * Hh + hh) * DVv + kd) * Nn + ns] = val;
        }
      }
}

// flash attention: one wave = 16 query rows x full DV, online softmax over keys
__global__ __launch_bounds__(128) void attn_kernel(
    const bf16* __restrict__ q, const bf16* __restrict__ k,
    const bf16* __restrict__ vt, const int* __restrict__ mask,
    bf16* __restrict__ o) {
  __shared__ __align__(32) bf16 plds[4][16 * 32];
  const int warp = threadIdx.x >> 5;
  const int lane = threadIdx.x & 31;
  const int wid  = blockIdx.x * 4 + warp;
  const int mt = wid & 63;
  const int h  = (wid >> 6) & 15;
  const int b  = wid >> 10;
  const bf16* qp = q  + (size_t)(b * Hh + h) * Nn * DKk;
  const bf16* kp = k  + (size_t)(b * Hh + h) * Nn * DKk;
  const bf16* vp = vt + (size_t)(b * Hh + h) * DVv * Nn;
  const int mbase = mt * 16;
  const int roff  = (lane & 16) ? 8 : 0;
  const int ncol  = lane & 15;

  v16bf qa0 = load_frag_a(qp, DKk, mbase, 0);
  v16bf qa1 = load_frag_a(qp, DKk, mbase, 32);

  v8f zero = {0.f,0.f,0.f,0.f,0.f,0.f,0.f,0.f};
  v8f accO[4] = {zero, zero, zero, zero};
  float mrow[8], lrow[8];
#pragma unroll
  for (int r = 0; r < 8; ++r) { mrow[r] = -3.0e38f; lrow[r] = 0.f; }

  const int* mp = mask + (size_t)b * Nn * Nn;

  for (int jb = 0; jb < Nn / 32; ++jb) {
    const int key0 = jb * 32;
    v8f s0 = zero, s1 = zero;
    {
      v16bf kb0 = load_frag_b(kp, DKk, key0,      0);
      v16bf kb1 = load_frag_b(kp, DKk, key0,      32);
      v16bf kb2 = load_frag_b(kp, DKk, key0 + 16, 0);
      v16bf kb3 = load_frag_b(kp, DKk, key0 + 16, 32);
      s0 = wmma_bf16(qa0, kb0, s0);
      s0 = wmma_bf16(qa1, kb1, s0);
      s1 = wmma_bf16(qa0, kb2, s1);
      s1 = wmma_bf16(qa1, kb3, s1);
    }
#pragma unroll
    for (int r = 0; r < 8; ++r) {
      int srow = mbase + r + roff;
      if (mp[(size_t)srow * Nn + key0 + ncol] == 0)      s0[r] = -1e10f;
      if (mp[(size_t)srow * Nn + key0 + 16 + ncol] == 0) s1[r] = -1e10f;
    }
    // online softmax: row r lives in lanes 0-15 (and row 8+r in lanes 16-31)
#pragma unroll
    for (int r = 0; r < 8; ++r) {
      float rm = fmaxf(s0[r], s1[r]);
      rm = fmaxf(rm, __shfl_xor(rm, 1, 32));
      rm = fmaxf(rm, __shfl_xor(rm, 2, 32));
      rm = fmaxf(rm, __shfl_xor(rm, 4, 32));
      rm = fmaxf(rm, __shfl_xor(rm, 8, 32));
      float nm   = fmaxf(mrow[r], rm);
      float corr = __expf(mrow[r] - nm);
      float p0 = __expf(s0[r] - nm);
      float p1 = __expf(s1[r] - nm);
      float ps = p0 + p1;
      ps += __shfl_xor(ps, 1, 32);
      ps += __shfl_xor(ps, 2, 32);
      ps += __shfl_xor(ps, 4, 32);
      ps += __shfl_xor(ps, 8, 32);
      lrow[r] = lrow[r] * corr + ps;
      mrow[r] = nm;
#pragma unroll
      for (int t2 = 0; t2 < 4; ++t2) accO[t2][r] *= corr;
      s0[r] = p0; s1[r] = p1;
    }
    // C-layout -> A-layout shuffle of probabilities through LDS
    bf16* pl = &plds[warp][0];
#pragma unroll
    for (int r = 0; r < 8; ++r) {
      int lr = r + roff;
      pl[lr * 32 + ncol]      = (bf16)s0[r];
      pl[lr * 32 + 16 + ncol] = (bf16)s1[r];
    }
    asm volatile("s_wait_dscnt 0" ::: "memory");
    v16bf pa = load_frag_a(pl, 32, 0, 0);
#pragma unroll
    for (int t2 = 0; t2 < 4; ++t2) {
      v16bf vb = load_frag_b(vp, Nn, t2 * 16, key0);   // Vt[dv][n], contiguous
      accO[t2] = wmma_bf16(pa, vb, accO[t2]);
    }
    asm volatile("s_wait_dscnt 0" ::: "memory");
  }
#pragma unroll
  for (int r = 0; r < 8; ++r) {
    float inv = 1.f / lrow[r];
    int srow = mbase + r + roff;
    bf16* op = o + (size_t)(b * Nn + srow) * Dd + h * DVv;
#pragma unroll
    for (int t2 = 0; t2 < 4; ++t2)
      op[t2 * 16 + ncol] = (bf16)(accO[t2][r] * inv);
  }
}

// h2 = h + o @ Wo^T
__global__ __launch_bounds__(128) void gemm_wo_kernel(
    const bf16* __restrict__ o, const bf16* __restrict__ wo,
    const float* __restrict__ h, float* __restrict__ h2) {
  const int wid = blockIdx.x * 4 + (threadIdx.x >> 5);
  const int tm = wid & 127, tn = wid >> 7;
  const int mb = tm * 32, nb = tn * 64;
  Acc acc;
  wave_gemm(o, Dd, wo, Dd, Dd, mb, nb, acc);
  const int lane = threadIdx.x & 31;
  const int roff = (lane & 16) ? 8 : 0, ncol = lane & 15;
#pragma unroll
  for (int i = 0; i < 2; ++i)
#pragma unroll
    for (int j = 0; j < 4; ++j)
#pragma unroll
      for (int r = 0; r < 8; ++r) {
        size_t idx = (size_t)(mb + i * 16 + r + roff) * Dd + nb + j * 16 + ncol;
        h2[idx] = h[idx] + acc.c[i][j][r];
      }
}

// u = gelu(r @ W1^T + b1)
__global__ __launch_bounds__(128) void gemm_ffn1_kernel(
    const bf16* __restrict__ rA, const bf16* __restrict__ w1,
    const float* __restrict__ b1, bf16* __restrict__ u) {
  const int wid = blockIdx.x * 4 + (threadIdx.x >> 5);
  const int tm = wid & 127, tn = wid >> 7;
  const int mb = tm * 32, nb = tn * 64;
  Acc acc;
  wave_gemm(rA, Dd, w1, Dd, Dd, mb, nb, acc);
  const int lane = threadIdx.x & 31;
  const int roff = (lane & 16) ? 8 : 0, ncol = lane & 15;
#pragma unroll
  for (int i = 0; i < 2; ++i)
#pragma unroll
    for (int j = 0; j < 4; ++j)
#pragma unroll
      for (int r = 0; r < 8; ++r) {
        int row = mb + i * 16 + r + roff;
        int col = nb + j * 16 + ncol;
        float x = acc.c[i][j][r] + b1[col];
        float g = 0.5f * x * (1.f + erff(x * 0.7071067811865476f));
        u[(size_t)row * DIi + col] = (bf16)g;
      }
}

// out = h2 + u @ W2^T + b2
__global__ __launch_bounds__(128) void gemm_ffn2_kernel(
    const bf16* __restrict__ u, const bf16* __restrict__ w2,
    const float* __restrict__ b2, const float* __restrict__ h2,
    float* __restrict__ out) {
  const int wid = blockIdx.x * 4 + (threadIdx.x >> 5);
  const int tm = wid & 127, tn = wid >> 7;
  const int mb = tm * 32, nb = tn * 64;
  Acc acc;
  wave_gemm(u, DIi, w2, DIi, DIi, mb, nb, acc);
  const int lane = threadIdx.x & 31;
  const int roff = (lane & 16) ? 8 : 0, ncol = lane & 15;
#pragma unroll
  for (int i = 0; i < 2; ++i)
#pragma unroll
    for (int j = 0; j < 4; ++j)
#pragma unroll
      for (int r = 0; r < 8; ++r) {
        int row = mb + i * 16 + r + roff;
        int col = nb + j * 16 + ncol;
        size_t idx = (size_t)row * Dd + col;
        out[idx] = h2[idx] + acc.c[i][j][r] + b2[col];
      }
}

// ------------------------------------------------------------------- driver
extern "C" void kernel_launch(void* const* d_in, const int* in_sizes, int n_in,
                              void* d_out, int out_size, void* d_ws, size_t ws_size,
                              hipStream_t stream) {
  (void)in_sizes; (void)n_in; (void)out_size; (void)ws_size;
  const float* x         = (const float*)d_in[0];
  const int*   mask      = (const int*)d_in[1];
  const float* Wq        = (const float*)d_in[2];
  const float* Wk        = (const float*)d_in[3];
  const float* Wv        = (const float*)d_in[4];
  const float* Wo        = (const float*)d_in[5];
  const float* ln_attn_g = (const float*)d_in[6];
  const float* ln_attn_b = (const float*)d_in[7];
  const float* ln_ffn_g  = (const float*)d_in[8];
  const float* ln_ffn_b  = (const float*)d_in[9];
  const float* ffn_ln_g  = (const float*)d_in[10];
  const float* ffn_ln_b  = (const float*)d_in[11];
  const float* W1        = (const float*)d_in[12];
  const float* b1        = (const float*)d_in[13];
  const float* W2        = (const float*)d_in[14];
  const float* b2        = (const float*)d_in[15];

  char* p = (char*)d_ws;
  auto take = [&](size_t bytes) -> char* {
    char* r = p;
    p += (bytes + 255) & ~(size_t)255;
    return r;
  };
  const size_t MN = (size_t)Bb * Nn;  // 4096 rows
  bf16* wqkv = (bf16*)take((size_t)3 * Hh * DKk * Dd * sizeof(bf16));
  bf16* wo   = (bf16*)take((size_t)Dd * Hh * DVv * sizeof(bf16));
  bf16* w1   = (bf16*)take((size_t)DIi * Dd * sizeof(bf16));
  bf16* w2   = (bf16*)take((size_t)Dd * DIi * sizeof(bf16));
  bf16* abuf = (bf16*)take(MN * Dd * sizeof(bf16));
  bf16* qb   = (bf16*)take(MN * Dd * sizeof(bf16));
  bf16* kb   = (bf16*)take(MN * Dd * sizeof(bf16));
  bf16* vtb  = (bf16*)take(MN * Dd * sizeof(bf16));
  bf16* ob   = (bf16*)take(MN * Dd * sizeof(bf16));
  bf16* rb   = (bf16*)take(MN * Dd * sizeof(bf16));
  bf16* ub   = (bf16*)take(MN * (size_t)DIi * sizeof(bf16));
  float* hbuf  = (float*)take(MN * Dd * sizeof(float));
  float* h2buf = (float*)take(MN * Dd * sizeof(float));
  float* hcar  = (float*)take(MN * Dd * sizeof(float));

  // weight conversion every call (deterministic; 1/sqrt(DK) folded into Wq)
  const int nQ = Hh * DKk * Dd;  // 1,048,576
  cvt_kernel<<<(nQ + 255) / 256, 256, 0, stream>>>(Wq, wqkv,          nQ, 0.125f);
  cvt_kernel<<<(nQ + 255) / 256, 256, 0, stream>>>(Wk, wqkv + nQ,     nQ, 1.0f);
  cvt_kernel<<<(nQ + 255) / 256, 256, 0, stream>>>(Wv, wqkv + 2 * nQ, nQ, 1.0f);
  cvt_kernel<<<(Dd * Hh * DVv + 255) / 256, 256, 0, stream>>>(Wo, wo, Dd * Hh * DVv, 1.0f);
  cvt_kernel<<<(DIi * Dd + 255) / 256, 256, 0, stream>>>(W1, w1, DIi * Dd, 1.0f);
  cvt_kernel<<<(Dd * DIi + 255) / 256, 256, 0, stream>>>(W2, w2, Dd * DIi, 1.0f);

  for (int t = 0; t < Tt; ++t) {
    const float* carry = (t == 0) ? x : hcar;
    pe_ln_kernel<<<4096, 256, 0, stream>>>(carry, ln_attn_g, ln_attn_b, hbuf, abuf, t);
    gemm_qkv_kernel<<<1536, 128, 0, stream>>>(abuf, wqkv, qb, kb, vtb);   // 128x48 tiles
    attn_kernel<<<1024, 128, 0, stream>>>(qb, kb, vtb, mask, ob);         // 4096 waves
    gemm_wo_kernel<<<512, 128, 0, stream>>>(ob, wo, hbuf, h2buf);         // 128x16 tiles
    double_ln_kernel<<<4096, 256, 0, stream>>>(h2buf, ln_ffn_g, ln_ffn_b,
                                               ffn_ln_g, ffn_ln_b, rb);
    gemm_ffn1_kernel<<<2048, 128, 0, stream>>>(rb, w1, b1, ub);           // 128x64 tiles
    gemm_ffn2_kernel<<<512, 128, 0, stream>>>(ub, w2, b2, h2buf,
                                              (t == Tt - 1) ? (float*)d_out : hcar);
  }
}
